// LSTM_GAT_Model_78640851189810
// MI455X (gfx1250) — matrix operand order repaired and checked by reading, not compiled, not run on hardware
//
#include <hip/hip_runtime.h>

typedef __attribute__((ext_vector_type(16))) _Float16 v16h;
typedef __attribute__((ext_vector_type(8)))  _Float16 v8h;
typedef __attribute__((ext_vector_type(8)))  float    v8f;
typedef __attribute__((ext_vector_type(4)))  float    v4f;
typedef __attribute__((ext_vector_type(4)))  unsigned int u32x4;
typedef __attribute__((ext_vector_type(8)))  int      i32x8;
typedef __attribute__((ext_vector_type(4)))  int      i32x4;

#define DEVFN static __device__ __forceinline__

#if defined(__has_builtin)
#if __has_builtin(__builtin_amdgcn_tensor_load_to_lds)
#define HAVE_TDM 1
#endif
#endif

// ---------------------------------------------------------------- WMMA core
DEVFN v8f wmma(v16h a, v16h b, v8f c) {
  return __builtin_amdgcn_wmma_f32_16x16x32_f16(false, a, false, b, (short)0, c,
                                                false, false);
}

DEVFN v8f zero8() {
  v8f z;
  #pragma unroll
  for (int i = 0; i < 8; ++i) z[i] = 0.f;
  return z;
}

DEVFN v16h cat16(v8h lo, v8h hi) {
  v16h r;
  #pragma unroll
  for (int i = 0; i < 8; ++i) { r[i] = lo[i]; r[8 + i] = hi[i]; }
  return r;
}

// A operand (16x32 f16): lane m=lane&15. lanes 0-15: K {0..7,16..23};
// lanes 16-31: K {8..15,24..31}.  row points at element (m, kbase).
DEVFN v16h loadA_f16(const _Float16* row, int lane) {
  int off = (lane >> 4) * 8;
  return cat16(*(const v8h*)(row + off), *(const v8h*)(row + 16 + off));
}

DEVFN v8h cvt8(const float* p) {
  v4f a = *(const v4f*)p, b = *(const v4f*)(p + 4);
  v8h r;
  #pragma unroll
  for (int i = 0; i < 4; ++i) { r[i] = (_Float16)a[i]; r[4 + i] = (_Float16)b[i]; }
  return r;
}
DEVFN v16h loadA_f32(const float* row, int lane) {
  int off = (lane >> 4) * 8;
  return cat16(cvt8(row + off), cvt8(row + 16 + off));
}

// B operand (32x16 f16): lane n=lane&15. lanes 0-15: K {0..7,16..23};
// lanes 16-31: K {8..15,24..31}.  base points at (kbase, ncol0).
DEVFN v16h loadB(const _Float16* base, int lane, int ldn) {
  const _Float16* p = base + (lane & 15) + ((lane >> 4) * 8) * ldn;
  v16h r;
  #pragma unroll
  for (int i = 0; i < 8; ++i) { r[i] = p[i * ldn]; r[8 + i] = p[(16 + i) * ldn]; }
  return r;
}

// ------------------------------------------------- Tensor Data Mover staging
#ifdef HAVE_TDM
DEVFN void tdm_load_2d_f16(void* lds_dst, const _Float16* gsrc,
                           unsigned rows, unsigned cols) {
  unsigned lds_off = (unsigned)(size_t)lds_dst;   // generic addr low 32 = LDS offset
  unsigned long long ga = (unsigned long long)(size_t)gsrc;
  u32x4 g0;
  g0.x = 1u;                                   // count=1, user descriptor
  g0.y = lds_off;                              // lds_addr
  g0.z = (unsigned)ga;                         // global_addr[31:0]
  g0.w = (unsigned)((ga >> 32) & 0x01FFFFFFull) | (2u << 30); // addr[56:32], type=2
  i32x8 g1;
  g1[0] = (1 << 16);                                            // data_size=2B
  g1[1] = (int)((cols & 0xFFFFu) << 16);                        // tensor_dim0 lo
  g1[2] = (int)((cols >> 16) & 0xFFFFu) | (int)((rows & 0xFFFFu) << 16); // dim0 hi|dim1 lo
  g1[3] = (int)((rows >> 16) & 0xFFFFu) | (int)((cols & 0xFFFFu) << 16); // dim1 hi|tile_dim0
  g1[4] = (int)(rows & 0xFFFFu);                                // tile_dim1, tile_dim2=0
  g1[5] = (int)cols;                                            // dim0_stride lo32
  g1[6] = 0;
  g1[7] = 0;
  i32x4 g2 = {0, 0, 0, 0};
  i32x4 g3 = {0, 0, 0, 0};
  i32x8 g4 = {0, 0, 0, 0, 0, 0, 0, 0};
  __builtin_amdgcn_tensor_load_to_lds(g0, g1, g2, g3, g4, 0);
}
#endif

DEVFN void stage2d(_Float16* sdst, const _Float16* gsrc, unsigned rows,
                   unsigned cols, int tid, int nthr, int wv, int issue_wv) {
#ifdef HAVE_TDM
  (void)tid; (void)nthr;
  if (wv == issue_wv) {
    tdm_load_2d_f16(sdst, gsrc, rows, cols);
    __builtin_amdgcn_s_wait_tensorcnt(0);
  }
#else
  (void)wv; (void)issue_wv;
  int count = (int)(rows * cols);
  for (int i = tid; i < count; i += nthr) sdst[i] = gsrc[i];
#endif
}

// -------------------------------------------------------------- LSTM kernel
// One block = 16 nodes for the full T=30 sequence. 4 waves; wave w owns gate
// columns [64w, 64w+64). Weight B-tiles are hoisted into VGPRs before the
// time loop, so the serial recurrence body is pure load-A -> v_wmma ->
// LDS gate exchange -> cell update.
template <bool XF32, int KIN, bool STORE_SEQ>
__global__ void lstm_kernel(const void* __restrict__ xin,
                            const _Float16* __restrict__ WihH,
                            const _Float16* __restrict__ WhhH,
                            const float* __restrict__ bih,
                            const float* __restrict__ bhh,
                            _Float16* __restrict__ hseq,   // N*T*64
                            _Float16* __restrict__ hlast,  // N*64
                            int n) {
  extern __shared__ unsigned char smem[];
  _Float16* sWih = (_Float16*)smem;               // KIN x 256
  _Float16* sWhh = sWih + KIN * 256;              // 64  x 256
  _Float16* sh   = sWhh + 64 * 256;               // 16  x 64
  float* sgate   = (float*)(sh + 16 * 64);        // 16  x 256
  float* sbias   = sgate + 16 * 256;              // 256

  const int tid = threadIdx.x, lane = tid & 31, wv = tid >> 5;
  const int node0 = blockIdx.x * 16;

  stage2d(sWih, WihH, KIN, 256, tid, 128, wv, 0);
  stage2d(sWhh, WhhH, 64, 256, tid, 128, wv, 1);
  for (int i = tid; i < 256; i += 128) sbias[i] = bih[i] + bhh[i];
  for (int i = tid; i < 16 * 64; i += 128) sh[i] = (_Float16)0.f;
  __syncthreads();

  const int r = tid & 15, cb = (tid >> 4) * 8;    // cell-update ownership
  float creg[8];
  #pragma unroll
  for (int j = 0; j < 8; ++j) creg[j] = 0.f;

  const int m = lane & 15;
  const int mrow = (node0 + m < n) ? (node0 + m) : (n - 1);
  const int col0 = wv * 64;

  // hoist loop-invariant weight B-tiles into registers
  v16h Bih[KIN / 32][4];
  #pragma unroll
  for (int ks = 0; ks < KIN / 32; ++ks)
    #pragma unroll
    for (int j = 0; j < 4; ++j)
      Bih[ks][j] = loadB(sWih + (ks * 32) * 256 + col0 + j * 16, lane, 256);
  v16h Bhh[2][4];
  #pragma unroll
  for (int ks = 0; ks < 2; ++ks)
    #pragma unroll
    for (int j = 0; j < 4; ++j)
      Bhh[ks][j] = loadB(sWhh + (ks * 32) * 256 + col0 + j * 16, lane, 256);

  for (int t = 0; t < 30; ++t) {
    v8f acc[4];
    #pragma unroll
    for (int j = 0; j < 4; ++j) acc[j] = zero8();

    // x(t) @ Wih
    #pragma unroll
    for (int ks = 0; ks < KIN / 32; ++ks) {
      v16h a;
      if (XF32)
        a = loadA_f32((const float*)xin + ((size_t)mrow * 30 + t) * KIN + ks * 32, lane);
      else
        a = loadA_f16((const _Float16*)xin + ((size_t)mrow * 30 + t) * KIN + ks * 32, lane);
      #pragma unroll
      for (int j = 0; j < 4; ++j) acc[j] = wmma(a, Bih[ks][j], acc[j]);
    }
    // h(t-1) @ Whh  (K = 64)
    #pragma unroll
    for (int ks = 0; ks < 2; ++ks) {
      v16h a = loadA_f16(sh + m * 64 + ks * 32, lane);
      #pragma unroll
      for (int j = 0; j < 4; ++j) acc[j] = wmma(a, Bhh[ks][j], acc[j]);
    }
    // scatter gate tile to LDS (D layout: VGPR g -> row 8*(lane>>4)+g, col = lane&15)
    const int rb = (lane >> 4) * 8;
    #pragma unroll
    for (int j = 0; j < 4; ++j) {
      int c = col0 + j * 16 + (lane & 15);
      #pragma unroll
      for (int g = 0; g < 8; ++g) sgate[(rb + g) * 256 + c] = acc[j][g];
    }
    __syncthreads();

    // cell update: thread owns (row r, cols cb..cb+7)
    #pragma unroll
    for (int j = 0; j < 8; ++j) {
      int c = cb + j;
      float gi = sgate[r * 256 + c]       + sbias[c];
      float gf = sgate[r * 256 + 64 + c]  + sbias[64 + c];
      float gg = sgate[r * 256 + 128 + c] + sbias[128 + c];
      float go = sgate[r * 256 + 192 + c] + sbias[192 + c];
      gi = 1.f / (1.f + __expf(-gi));
      gf = 1.f / (1.f + __expf(-gf));
      go = 1.f / (1.f + __expf(-go));
      gg = tanhf(gg);
      creg[j] = gf * creg[j] + gi * gg;
      float hv = go * tanhf(creg[j]);
      sh[r * 64 + c] = (_Float16)hv;
      if (STORE_SEQ) {
        if (node0 + r < n)
          hseq[((size_t)(node0 + r) * 30 + t) * 64 + c] = (_Float16)hv;
      } else if (t == 29) {
        if (node0 + r < n)
          hlast[(size_t)(node0 + r) * 64 + c] = (_Float16)hv;
      }
    }
    __syncthreads();
  }
}

// ------------------------------------------------------- dense GEMM (WMMA)
template <int K, int WAVES, int NT>
__global__ void gemm_f16_kernel(const _Float16* __restrict__ A,  // n x K
                                const _Float16* __restrict__ W,  // K x Ntot f16
                                float* __restrict__ out,         // n x Ntot
                                int n) {
  const int Ntot = WAVES * NT * 16;
  extern __shared__ unsigned char smem[];
  _Float16* sW = (_Float16*)smem;
  const int tid = threadIdx.x, lane = tid & 31, wv = tid >> 5;
  const int node0 = blockIdx.x * 16;
  stage2d(sW, W, K, (unsigned)Ntot, tid, WAVES * 32, wv, 0);
  __syncthreads();

  const int m = lane & 15;
  const int mrow = (node0 + m < n) ? (node0 + m) : (n - 1);

  v16h B[K / 32][NT];
  #pragma unroll
  for (int ks = 0; ks < K / 32; ++ks)
    #pragma unroll
    for (int j = 0; j < NT; ++j)
      B[ks][j] = loadB(sW + (ks * 32) * Ntot + wv * NT * 16 + j * 16, lane, Ntot);

  v8f acc[NT];
  #pragma unroll
  for (int j = 0; j < NT; ++j) acc[j] = zero8();

  #pragma unroll
  for (int ks = 0; ks < K / 32; ++ks) {
    v16h a = loadA_f16(A + (size_t)mrow * K + ks * 32, lane);
    #pragma unroll
    for (int j = 0; j < NT; ++j) acc[j] = wmma(a, B[ks][j], acc[j]);
  }
  const int rb = (lane >> 4) * 8;
  #pragma unroll
  for (int j = 0; j < NT; ++j) {
    int c = wv * NT * 16 + j * 16 + (lane & 15);
    #pragma unroll
    for (int g = 0; g < 8; ++g) {
      int row = node0 + rb + g;
      if (row < n) out[(size_t)row * Ntot + c] = acc[j][g];
    }
  }
}

// ---------------------------------------------------------- small utilities
__global__ void cvt_f16_kernel(const float* __restrict__ s, _Float16* __restrict__ d, int cnt) {
  int i = blockIdx.x * blockDim.x + threadIdx.x;
  if (i < cnt) d[i] = (_Float16)s[i];
}
__global__ void zero_f32_kernel(float* p, int cnt) {
  int i = blockIdx.x * blockDim.x + threadIdx.x;
  if (i < cnt) p[i] = 0.f;
}
__global__ void zero_u32_kernel(unsigned* p, int cnt) {
  int i = blockIdx.x * blockDim.x + threadIdx.x;
  if (i < cnt) p[i] = 0u;
}
__global__ void bias_bcast_kernel(float* o, const float* b, int total, int C) {
  int i = blockIdx.x * blockDim.x + threadIdx.x;
  if (i < total) o[i] = b[i % C];
}
__global__ void relu_bias_cvt_kernel(const float* in, const float* b,
                                     _Float16* oh, int total, int C) {
  int i = blockIdx.x * blockDim.x + threadIdx.x;
  if (i >= total) return;
  float v = in[i] + b[i % C];
  oh[i] = (_Float16)(v > 0.f ? v : 0.f);
}
__global__ void att_dots_kernel(const float* __restrict__ feat,
                                const float* __restrict__ atts,
                                const float* __restrict__ attd,
                                float* as_, float* ad_, int n, int Hh, int C) {
  int idx = blockIdx.x * blockDim.x + threadIdx.x;
  if (idx >= n * Hh) return;
  int node = idx / Hh, hd = idx % Hh;
  const float* f = feat + (size_t)node * Hh * C + hd * C;
  float s = 0.f, d = 0.f;
  for (int c = 0; c < C; ++c) { float v = f[c]; s += v * atts[hd * C + c]; d += v * attd[hd * C + c]; }
  as_[idx] = s;
  ad_[idx] = d;
}

// ------------------------------------------------------------- edge kernels
DEVFN unsigned fkey(float x) {
  unsigned u = __float_as_uint(x);
  return (u & 0x80000000u) ? ~u : (u | 0x80000000u);
}
DEVFN float funkey(unsigned u) {
  unsigned b = (u & 0x80000000u) ? (u ^ 0x80000000u) : ~u;
  return __uint_as_float(b);
}
DEVFN void edge_sd(const long long* ei, long long E, long long e,
                   long long& s, long long& d) {
  if (e < E) { s = ei[e]; d = ei[E + e]; } else { s = d = e - E; }
}

__global__ void edge_logits_max_kernel(const long long* __restrict__ ei, long long E, int n,
                                       const float* __restrict__ asrc,
                                       const float* __restrict__ adst,
                                       float* __restrict__ logits,
                                       unsigned* __restrict__ mkey, int Hh) {
  long long idx = blockIdx.x * (long long)blockDim.x + threadIdx.x;
  long long tot = (E + n) * Hh;
  if (idx >= tot) return;
  long long e = idx / Hh; int hd = (int)(idx % Hh);
  long long s, d; edge_sd(ei, E, e, s, d);
  float l = asrc[s * Hh + hd] + adst[d * Hh + hd];
  l = l > 0.f ? l : 0.2f * l;            // leaky_relu 0.2
  logits[idx] = l;
  atomicMax(&mkey[d * Hh + hd], fkey(l));
}

__global__ void edge_expsum_kernel(const long long* __restrict__ ei, long long E, int n,
                                   float* __restrict__ logits,
                                   const unsigned* __restrict__ mkey,
                                   float* __restrict__ ssum, int Hh) {
  long long idx = blockIdx.x * (long long)blockDim.x + threadIdx.x;
  long long tot = (E + n) * Hh;
  if (idx >= tot) return;
  long long e = idx / Hh; int hd = (int)(idx % Hh);
  long long s, d; edge_sd(ei, E, e, s, d);
  float ev = __expf(logits[idx] - funkey(mkey[d * Hh + hd]));
  logits[idx] = ev;
  atomicAdd(&ssum[d * Hh + hd], ev);
}

__global__ void edge_accum_kernel(const long long* __restrict__ ei, long long E, int n,
                                  const float* __restrict__ logits,
                                  const float* __restrict__ ssum,
                                  const float* __restrict__ feat,
                                  float* __restrict__ out, int Hh, int C) {
  long long idx = blockIdx.x * (long long)blockDim.x + threadIdx.x;
  long long tot = (E + n) * Hh;
  if (idx >= tot) return;
  long long e = idx / Hh; int hd = (int)(idx % Hh);
  long long s, d; edge_sd(ei, E, e, s, d);
  float alpha = logits[idx] / ssum[d * Hh + hd];
  const float* fs = feat + (size_t)s * Hh * C + hd * C;
  float* od = out + (size_t)d * Hh * C + hd * C;
  for (int c = 0; c < C; ++c) atomicAdd(&od[c], alpha * fs[c]);
}

// --------------------------------------------------------------- launcher
extern "C" void kernel_launch(void* const* d_in, const int* in_sizes, int n_in,
                              void* d_out, int out_size, void* d_ws, size_t ws_size,
                              hipStream_t stream) {
  const int T = 30, IN = 32, H = 64, G = 256, HEADS = 4, OUTC = 32;
  const float* x      = (const float*)d_in[0];
  const long long* ei = (const long long*)d_in[1];
  const float* Wih0 = (const float*)d_in[2];
  const float* Whh0 = (const float*)d_in[3];
  const float* bih0 = (const float*)d_in[4];
  const float* bhh0 = (const float*)d_in[5];
  const float* Wih1 = (const float*)d_in[6];
  const float* Whh1 = (const float*)d_in[7];
  const float* bih1 = (const float*)d_in[8];
  const float* bhh1 = (const float*)d_in[9];
  const float* W1   = (const float*)d_in[10];
  const float* as1  = (const float*)d_in[11];
  const float* ad1  = (const float*)d_in[12];
  const float* b1   = (const float*)d_in[13];
  const float* W2   = (const float*)d_in[14];
  const float* as2  = (const float*)d_in[15];
  const float* ad2  = (const float*)d_in[16];
  const float* b2   = (const float*)d_in[17];
  float* out = (float*)d_out;

  const int n = in_sizes[0] / (T * IN);
  const long long E = in_sizes[1] / 2;

  unsigned char* ws = (unsigned char*)d_ws;
  size_t off = 0;
  auto carve = [&](size_t bytes) -> void* {
    void* p = ws + off;
    off = (off + bytes + 255) & ~(size_t)255;
    return p;
  };
  _Float16* Wih0h = (_Float16*)carve((size_t)IN * G * 2);
  _Float16* Whh0h = (_Float16*)carve((size_t)H * G * 2);
  _Float16* Wih1h = (_Float16*)carve((size_t)H * G * 2);
  _Float16* Whh1h = (_Float16*)carve((size_t)H * G * 2);
  _Float16* W1h   = (_Float16*)carve((size_t)H * (HEADS * H) * 2);
  _Float16* W2h   = (_Float16*)carve((size_t)(HEADS * H) * OUTC * 2);
  _Float16* h0seq = (_Float16*)carve((size_t)n * T * H * 2);
  _Float16* hlast = (_Float16*)carve((size_t)n * H * 2);
  float*    feat1 = (float*)carve((size_t)n * 256 * 4);
  float*    asrc1 = (float*)carve((size_t)n * HEADS * 4);
  float*    adst1 = (float*)carve((size_t)n * HEADS * 4);
  unsigned* mkey1 = (unsigned*)carve((size_t)n * HEADS * 4);
  float*    ssum1 = (float*)carve((size_t)n * HEADS * 4);
  float*    lgt1  = (float*)carve((size_t)(E + n) * HEADS * 4);
  float*    out1  = (float*)carve((size_t)n * 256 * 4);
  _Float16* out1h = (_Float16*)carve((size_t)n * 256 * 2);
  float*    feat2 = (float*)carve((size_t)n * OUTC * 4);
  float*    asrc2 = (float*)carve((size_t)n * 4);
  float*    adst2 = (float*)carve((size_t)n * 4);
  unsigned* mkey2 = (unsigned*)carve((size_t)n * 4);
  float*    ssum2 = (float*)carve((size_t)n * 4);
  float*    lgt2  = (float*)carve((size_t)(E + n) * 4);
  (void)ws_size; (void)n_in; (void)out_size;

  auto blocks = [](long long t, int b) { return (unsigned)((t + b - 1) / b); };

  // f32 -> f16 weight staging
  cvt_f16_kernel<<<blocks(IN * G, 256), 256, 0, stream>>>(Wih0, Wih0h, IN * G);
  cvt_f16_kernel<<<blocks(H * G, 256), 256, 0, stream>>>(Whh0, Whh0h, H * G);
  cvt_f16_kernel<<<blocks(H * G, 256), 256, 0, stream>>>(Wih1, Wih1h, H * G);
  cvt_f16_kernel<<<blocks(H * G, 256), 256, 0, stream>>>(Whh1, Whh1h, H * G);
  cvt_f16_kernel<<<blocks(H * G, 256), 256, 0, stream>>>(W1, W1h, H * G);
  cvt_f16_kernel<<<blocks(G * OUTC, 256), 256, 0, stream>>>(W2, W2h, G * OUTC);

  // LSTM layers (fused time loop, TDM-staged weights, WMMA GEMMs)
  const unsigned gridN = blocks(n, 16);
  size_t shm0 = (size_t)(32 * 256 + 64 * 256 + 16 * 64) * 2 + (size_t)(16 * 256 + 256) * 4;
  size_t shm1 = (size_t)(64 * 256 + 64 * 256 + 16 * 64) * 2 + (size_t)(16 * 256 + 256) * 4;
  lstm_kernel<true, 32, true><<<gridN, 128, shm0, stream>>>(
      x, Wih0h, Whh0h, bih0, bhh0, h0seq, nullptr, n);
  lstm_kernel<false, 64, false><<<gridN, 128, shm1, stream>>>(
      h0seq, Wih1h, Whh1h, bih1, bhh1, nullptr, hlast, n);

  // ---- GAT layer 1 ----
  gemm_f16_kernel<64, 4, 4><<<gridN, 128, (size_t)64 * 256 * 2, stream>>>(hlast, W1h, feat1, n);
  att_dots_kernel<<<blocks((long long)n * HEADS, 256), 256, 0, stream>>>(
      feat1, as1, ad1, asrc1, adst1, n, HEADS, H);
  zero_u32_kernel<<<blocks((long long)n * HEADS, 256), 256, 0, stream>>>(mkey1, n * HEADS);
  zero_f32_kernel<<<blocks((long long)n * HEADS, 256), 256, 0, stream>>>(ssum1, n * HEADS);
  zero_f32_kernel<<<blocks((long long)n * 256, 256), 256, 0, stream>>>(out1, n * 256);
  const long long tot1 = (E + n) * HEADS;
  edge_logits_max_kernel<<<blocks(tot1, 256), 256, 0, stream>>>(ei, E, n, asrc1, adst1, lgt1, mkey1, HEADS);
  edge_expsum_kernel<<<blocks(tot1, 256), 256, 0, stream>>>(ei, E, n, lgt1, mkey1, ssum1, HEADS);
  edge_accum_kernel<<<blocks(tot1, 256), 256, 0, stream>>>(ei, E, n, lgt1, ssum1, feat1, out1, HEADS, H);
  relu_bias_cvt_kernel<<<blocks((long long)n * 256, 256), 256, 0, stream>>>(out1, b1, out1h, n * 256, 256);

  // ---- GAT layer 2 ----
  gemm_f16_kernel<256, 2, 1><<<gridN, 64, (size_t)256 * 32 * 2, stream>>>(out1h, W2h, feat2, n);
  att_dots_kernel<<<blocks(n, 256), 256, 0, stream>>>(feat2, as2, ad2, asrc2, adst2, n, 1, OUTC);
  zero_u32_kernel<<<blocks(n, 256), 256, 0, stream>>>(mkey2, n);
  zero_f32_kernel<<<blocks(n, 256), 256, 0, stream>>>(ssum2, n);
  bias_bcast_kernel<<<blocks((long long)n * OUTC, 256), 256, 0, stream>>>(out, b2, n * OUTC, OUTC);
  const long long tot2 = E + n;
  edge_logits_max_kernel<<<blocks(tot2, 256), 256, 0, stream>>>(ei, E, n, asrc2, adst2, lgt2, mkey2, 1);
  edge_expsum_kernel<<<blocks(tot2, 256), 256, 0, stream>>>(ei, E, n, lgt2, mkey2, ssum2, 1);
  edge_accum_kernel<<<blocks(tot2, 256), 256, 0, stream>>>(ei, E, n, lgt2, ssum2, feat2, out, 1, OUTC);
}
